// Model_76768245449416
// MI455X (gfx1250) — compile-verified, matrix operand order checked by reading, not run on hardware
//
#include <hip/hip_runtime.h>
#include <stdint.h>

// ---------------------------------------------------------------------------
// RPN head for MI455X (gfx1250): bf16 WMMA implicit-GEMM 3x3 conv (double-
// buffered LDS pipeline) + heads + exact top-k (bitonic) + bitmask NMS.
// wave32 everywhere.
// ---------------------------------------------------------------------------

typedef __attribute__((ext_vector_type(16))) __bf16        v16bf;
typedef __attribute__((ext_vector_type(8)))  float         v8f;
typedef __attribute__((ext_vector_type(4)))  unsigned int  u32x4;

#define BB     8
#define HH     32
#define WW     32
#define CC     512
#define COUT   512
#define NPIX   (BB*HH*WW)      // 8192
#define AA     15
#define NANCH  (HH*WW*AA)      // 15360 per image
#define NSORT  16384           // padded to pow2 for bitonic
#define KMAX   2000
#define NWORD  63              // ceil(2016/32) words of suppression bits
#define NSTAGE 72              // 4608 K / 64 per stage

static __device__ __forceinline__ float bf2f(unsigned short h) {
  unsigned int u = ((unsigned int)h) << 16;
  float f; __builtin_memcpy(&f, &u, 4); return f;
}
static __device__ __forceinline__ unsigned short f2bf(float f) {
  unsigned int u; __builtin_memcpy(&u, &f, 4);
  unsigned int r = u + 0x7FFFu + ((u >> 16) & 1u);   // round-to-nearest-even
  return (unsigned short)(r >> 16);
}

// ---------------- 1) feat NCHW f32 -> NHWC bf16 ----------------------------
__global__ void nhwc_kernel(const float* __restrict__ feat,
                            unsigned short* __restrict__ fn) {
  int gid = blockIdx.x * 256 + threadIdx.x;
  if (gid >= NPIX * CC) return;
  int c = gid & 511;
  int pix = gid >> 9;
  int x = pix & 31, y = (pix >> 5) & 31, b = pix >> 10;
  fn[gid] = f2bf(feat[(((size_t)(b * CC + c)) * HH + y) * WW + x]);
}

// ---------------- 2) conv_w OIHW f32 -> [tap][o][i] bf16 -------------------
__global__ void wt_kernel(const float* __restrict__ w,
                          unsigned short* __restrict__ wt) {
  int gid = blockIdx.x * 256 + threadIdx.x;
  if (gid >= 9 * COUT * CC) return;
  int c = gid & 511;
  int r = gid >> 9;          // t*512 + o
  int o = r & 511;
  int t = r >> 9;
  wt[gid] = f2bf(w[(size_t)(o * CC + c) * 9 + t]);
}

// ---------------- 3) 3x3 conv as implicit GEMM with bf16 WMMA --------------
// M = 8192 pixels, N = 512 Cout, K = 9*512.  Block tile 64x128, 8 waves (2x4),
// wave tile 32x32 = 2x2 WMMA 16x16x32_bf16 tiles, f32 accumulate.
// Software pipeline: register-staged ping-pong LDS double buffer, 1 barrier
// per K-stage; stage s+1 global loads issued before stage s WMMA work.
__global__ __launch_bounds__(256) void conv3x3_wmma(
    const unsigned short* __restrict__ fn, const unsigned short* __restrict__ wt,
    const float* __restrict__ conv_b, unsigned short* __restrict__ xo) {
  __shared__ unsigned short As[2][64 * 72];   // padded stride 72 (144B = 9*16B)
  __shared__ unsigned short Bs[2][128 * 72];

  const int tid  = threadIdx.x;
  const int lane = tid & 31, wv = tid >> 5;
  const int lm   = lane & 15, half = lane >> 4;
  const int mwave = (wv & 1) * 32;      // 2 waves along M
  const int nwave = (wv >> 1) * 32;     // 4 waves along N
  const int pm0  = blockIdx.x * 64;
  const int n0g  = blockIdx.y * 128;

  v8f acc[2][2] = {};
  const int ar  = tid >> 3;             // 0..31 (A row group)
  const int ach = tid & 7;              // 16B chunk within 64-K slice
  const int bn  = tid >> 3;             // B row group

  u32x4 aReg[2], bReg[4];

  // ---- stage-load into registers (A: zero-padded 3x3 halo; B: weights) ----
#define LOAD_STAGE(stage_)                                                     \
  {                                                                            \
    const int kk_ = (stage_) * 64;                                             \
    const int t_  = kk_ >> 9;          /* 512 % 64 == 0: never straddles */    \
    const int c0_ = kk_ & 511;                                                 \
    const int dy_ = t_ / 3 - 1, dx_ = t_ % 3 - 1;                              \
    _Pragma("unroll")                                                          \
    for (int rr = 0; rr < 2; ++rr) {                                           \
      int r_  = ar + rr * 32;                                                  \
      int pm_ = pm0 + r_;                                                      \
      int b_  = pm_ >> 10;                                                     \
      int y_  = (pm_ >> 5) & 31;                                               \
      int x_  = pm_ & 31;                                                      \
      int yy_ = y_ + dy_, xx_ = x_ + dx_;                                      \
      aReg[rr] = (u32x4){0, 0, 0, 0};                                          \
      if ((unsigned)yy_ < 32u && (unsigned)xx_ < 32u)                          \
        aReg[rr] = *(const u32x4*)(fn +                                        \
            (size_t)((((b_ << 5) + yy_) << 5) + xx_) * CC + c0_ + ach * 8);    \
    }                                                                          \
    _Pragma("unroll")                                                          \
    for (int rr = 0; rr < 4; ++rr) {                                           \
      int n_ = bn + rr * 32;                                                   \
      bReg[rr] = *(const u32x4*)(wt +                                          \
          ((size_t)(t_ * COUT + n0g + n_)) * CC + c0_ + ach * 8);              \
    }                                                                          \
  }

#define STORE_STAGE(buf_)                                                      \
  {                                                                            \
    _Pragma("unroll")                                                          \
    for (int rr = 0; rr < 2; ++rr)                                             \
      *(u32x4*)&As[buf_][(ar + rr * 32) * 72 + ach * 8] = aReg[rr];            \
    _Pragma("unroll")                                                          \
    for (int rr = 0; rr < 4; ++rr)                                             \
      *(u32x4*)&Bs[buf_][(bn + rr * 32) * 72 + ach * 8] = bReg[rr];            \
  }

  LOAD_STAGE(0);
  STORE_STAGE(0);
  __syncthreads();

  for (int stage = 0; stage < NSTAGE; ++stage) {
    const int buf = stage & 1;
    // issue next stage's global loads first; latency hidden under WMMA work
    if (stage + 1 < NSTAGE) {
      LOAD_STAGE(stage + 1);
      // prefetch stage+2 weights into caches (global_prefetch_b8)
      if (stage + 2 < NSTAGE) {
        int kk2 = (stage + 2) * 64, t2 = kk2 >> 9, c02 = kk2 & 511;
        __builtin_prefetch(wt + ((size_t)(t2 * COUT + n0g + bn)) * CC + c02, 0, 1);
      }
    }

    // compute current stage from LDS buffer `buf`
#pragma unroll
    for (int ks = 0; ks < 2; ++ks) {
      const int k0 = ks * 32;
      union { v16bf v; u32x4 q[2]; } af[2], bfr[2];
      // A frag (ISA 16-bit A layout: lanes 0-15 K{0..7,16..23}, 16-31 K{8..15,24..31})
#pragma unroll
      for (int mt = 0; mt < 2; ++mt) {
        int row = mwave + mt * 16 + lm;
        af[mt].q[0] = *(const u32x4*)&As[buf][row * 72 + k0 + half * 8];
        af[mt].q[1] = *(const u32x4*)&As[buf][row * 72 + k0 + 16 + half * 8];
      }
      // B frag: column-major B == rows of W^T, same K striping
#pragma unroll
      for (int nt = 0; nt < 2; ++nt) {
        int col = nwave + nt * 16 + lm;
        bfr[nt].q[0] = *(const u32x4*)&Bs[buf][col * 72 + k0 + half * 8];
        bfr[nt].q[1] = *(const u32x4*)&Bs[buf][col * 72 + k0 + 16 + half * 8];
      }
#pragma unroll
      for (int mt = 0; mt < 2; ++mt)
#pragma unroll
        for (int nt = 0; nt < 2; ++nt)
          acc[mt][nt] = __builtin_amdgcn_wmma_f32_16x16x32_bf16(
              false, af[mt].v, false, bfr[nt].v, (short)0, acc[mt][nt], false, false);
    }

    // commit staged registers to the other buffer, one barrier per stage
    if (stage + 1 < NSTAGE) {
      STORE_STAGE(1 - buf);
      __syncthreads();
    }
  }
#undef LOAD_STAGE
#undef STORE_STAGE

  // epilogue: bias + ReLU, per ISA f32 C layout (VGPR v -> M = v + 8*half)
#pragma unroll
  for (int mt = 0; mt < 2; ++mt)
#pragma unroll
    for (int nt = 0; nt < 2; ++nt) {
      int oc = n0g + nwave + nt * 16 + lm;
      float bias = conv_b[oc];
#pragma unroll
      for (int v = 0; v < 8; ++v) {
        int m = mwave + mt * 16 + half * 8 + v;
        float val = acc[mt][nt][v] + bias;
        val = val > 0.f ? val : 0.f;
        xo[(size_t)(pm0 + m) * COUT + oc] = f2bf(val);
      }
    }
}

// ---------------- 4) 1x1 heads (cls 15 + reg 60) ---------------------------
__global__ __launch_bounds__(256) void heads_kernel(
    const unsigned short* __restrict__ x, const float* __restrict__ cls_w,
    const float* __restrict__ cls_b, const float* __restrict__ reg_w,
    const float* __restrict__ reg_b, float* __restrict__ cls_o,
    float* __restrict__ reg_o) {
  __shared__ unsigned short xs[16 * 512];
  const int tid = threadIdx.x;
  const int pm0 = blockIdx.x * 16;
#pragma unroll
  for (int it = 0; it < 4; ++it) {
    int unit = tid + it * 256;
    int p = unit >> 6, ch = unit & 63;
    *(u32x4*)&xs[p * 512 + ch * 8] =
        *(const u32x4*)(x + (size_t)(pm0 + p) * 512 + ch * 8);
  }
  __syncthreads();
  for (int task = tid; task < 16 * 75; task += 256) {
    int p = task / 75, oc = task % 75;
    const float* wp; float s;
    if (oc < 15) { wp = cls_w + oc * 512; s = cls_b[oc]; }
    else         { wp = reg_w + (oc - 15) * 512; s = reg_b[oc - 15]; }
    const unsigned short* xp = &xs[p * 512];
#pragma unroll 8
    for (int k = 0; k < 512; ++k) s += wp[k] * bf2f(xp[k]);
    int pm = pm0 + p;
    if (oc < 15) cls_o[(size_t)pm * 15 + oc] = s;
    else         reg_o[(size_t)pm * 60 + (oc - 15)] = s;
  }
}

// ---------------- 5) decode + clip + filter + sort keys --------------------
__global__ void decode_kernel(const float* __restrict__ cls_o,
                              const float* __restrict__ reg_o,
                              float* __restrict__ boxes,
                              unsigned long long* __restrict__ keys) {
  int gid = blockIdx.x * 256 + threadIdx.x;
  if (gid >= BB * NSORT) return;
  int b = gid >> 14, i = gid & (NSORT - 1);
  unsigned long long key;
  if (i < NANCH) {
    int a = i % 15, pix = i / 15;
    int y = pix >> 5, x = pix & 31;
    int pm = (b << 10) + pix;
    float prob = 1.f / (1.f + expf(-cls_o[(size_t)pm * 15 + a]));
    const float sizes[5]  = {32.f, 64.f, 128.f, 256.f, 512.f};
    const float ratios[3] = {0.5f, 1.f, 2.f};
    float size = sizes[a % 5], ratio = ratios[a / 5];
    float hr = sqrtf(ratio), wr = 1.f / hr;
    float hs = hr * size, ws_ = wr * size;
    float ax1 = rintf(-ws_ * 0.5f), ay1 = rintf(-hs * 0.5f);
    float ax2 = rintf( ws_ * 0.5f), ay2 = rintf( hs * 0.5f);
    float sx = (float)(x * 16), sy = (float)(y * 16);
    float x1 = sx + ax1, y1 = sy + ay1, x2 = sx + ax2, y2 = sy + ay2;
    float w = x2 - x1 + 1.f, h = y2 - y1 + 1.f;
    float cx = x1 + 0.5f * w, cy = y1 + 0.5f * h;
    const float* dp = reg_o + (size_t)pm * 60 + a * 4;
    float pcx = dp[0] * w + cx, pcy = dp[1] * h + cy;
    float pw = expf(dp[2]) * w, ph = expf(dp[3]) * h;
    float bx1 = fminf(fmaxf(pcx - 0.5f * pw, 0.f), 512.f);
    float by1 = fminf(fmaxf(pcy - 0.5f * ph, 0.f), 512.f);
    float bx2 = fminf(fmaxf(pcx + 0.5f * pw, 0.f), 512.f);
    float by2 = fminf(fmaxf(pcy + 0.5f * ph, 0.f), 512.f);
    float* bo = boxes + ((size_t)b * NANCH + i) * 4;
    bo[0] = bx1; bo[1] = by1; bo[2] = bx2; bo[3] = by2;
    bool ok = (bx2 - bx1 >= 1.f) && (by2 - by1 >= 1.f);
    float s = ok ? prob : -__builtin_inff();
    unsigned int u; __builtin_memcpy(&u, &s, 4);
    unsigned int ss = (u & 0x80000000u) ? ~u : (u | 0x80000000u); // sortable
    key = ((unsigned long long)ss << 32) | (0xFFFFFFFFu - (unsigned)i);
  } else {
    key = ((unsigned long long)0x007FFFFFu << 32);  // -inf padding, sorts last
  }
  keys[(size_t)b * NSORT + i] = key;
}

// ---------------- 6) per-image descending bitonic sort ---------------------
__global__ __launch_bounds__(1024) void sort_kernel(unsigned long long* keys) {
  unsigned long long* kb = keys + (size_t)blockIdx.x * NSORT;
  const int tid = threadIdx.x;
  for (int k = 2; k <= NSORT; k <<= 1) {
    for (int j = k >> 1; j > 0; j >>= 1) {
      for (int e = 0; e < NSORT / 1024; ++e) {
        int i = e * 1024 + tid;
        int ixj = i ^ j;
        if (ixj > i) {
          unsigned long long a = kb[i], c = kb[ixj];
          bool desc = ((i & k) == 0);
          if (desc ? (a < c) : (a > c)) { kb[i] = c; kb[ixj] = a; }
        }
      }
      __syncthreads();
    }
  }
}

// ---------------- 7a) gather top-2000 --------------------------------------
__global__ void gather_kernel(const unsigned long long* __restrict__ keys,
                              const float* __restrict__ boxes,
                              float* __restrict__ gbox, float* __restrict__ gsc,
                              unsigned int* __restrict__ gval) {
  int b = blockIdx.x;
  for (int t = threadIdx.x; t < KMAX; t += 256) {
    unsigned long long key = keys[(size_t)b * NSORT + t];
    unsigned int idx = 0xFFFFFFFFu - (unsigned int)key;
    if (idx >= NANCH) idx = 0;
    unsigned int ssu = (unsigned int)(key >> 32);
    unsigned int u = (ssu & 0x80000000u) ? (ssu ^ 0x80000000u) : ~ssu;
    float s; __builtin_memcpy(&s, &u, 4);
    const float* bo = boxes + ((size_t)b * NANCH + idx) * 4;
    float* g = gbox + ((size_t)b * KMAX + t) * 4;
    g[0] = bo[0]; g[1] = bo[1]; g[2] = bo[2]; g[3] = bo[3];
    gsc[(size_t)b * KMAX + t]  = s;
    gval[(size_t)b * KMAX + t] = s > 0.5f ? 1u : 0u;
  }
}

// ---------------- 7b) 2000x2000 IoU suppression bitmask --------------------
__global__ void iou_kernel(const float* __restrict__ gbox,
                           unsigned int* __restrict__ mask) {
  int b = blockIdx.y;
  int idx = blockIdx.x * 256 + threadIdx.x;
  if (idx >= KMAX * NWORD) return;
  int i = idx / NWORD, wj = idx % NWORD;
  const float* gi = gbox + ((size_t)b * KMAX + i) * 4;
  float x1 = gi[0], y1 = gi[1], x2 = gi[2], y2 = gi[3];
  float ai = (x2 - x1) * (y2 - y1);
  unsigned int bits = 0;
  int jbase = wj * 32;
  for (int jj = 0; jj < 32; ++jj) {
    int j = jbase + jj;
    if (j > i && j < KMAX) {
      const float* gj = gbox + ((size_t)b * KMAX + j) * 4;
      float aj = (gj[2] - gj[0]) * (gj[3] - gj[1]);
      float iw = fmaxf(fminf(x2, gj[2]) - fmaxf(x1, gj[0]), 0.f);
      float ih = fmaxf(fminf(y2, gj[3]) - fmaxf(y1, gj[1]), 0.f);
      float inter = iw * ih;
      float uni = ai + aj - inter;
      float iou = uni > 0.f ? inter / uni : 0.f;
      if (iou > 0.7f) bits |= (1u << jj);
    }
  }
  mask[((size_t)b * KMAX + i) * NWORD + wj] = bits;
}

// ---------------- 7c) single-wave greedy reduce + output -------------------
__global__ __launch_bounds__(32) void nms_reduce_kernel(
    const unsigned int* __restrict__ mask, const unsigned int* __restrict__ gval,
    const float* __restrict__ gbox, const float* __restrict__ gsc,
    float* __restrict__ out) {
  __shared__ unsigned int remv[NWORD];
  __shared__ unsigned int kf[KMAX];
  int b = blockIdx.x, lane = threadIdx.x;
  for (int w = lane; w < NWORD; w += 32) remv[w] = 0;
  __syncthreads();
  for (int i = 0; i < KMAX; ++i) {
    unsigned int sup = (remv[i >> 5] >> (i & 31)) & 1u;
    unsigned int ki = (!sup) && gval[(size_t)b * KMAX + i];
    if (lane == 0) kf[i] = ki;
    if (ki) {
      const unsigned int* mr = mask + ((size_t)b * KMAX + i) * NWORD;
      for (int w = lane; w < NWORD; w += 32) remv[w] |= mr[w];
    }
  }
  __syncthreads();
  float* props = out;                  // (8,2000,4)
  float* sco   = out + BB * KMAX * 4;  // (8,2000)
  float* kee   = out + BB * KMAX * 5;  // (8,2000)
  for (int t = lane; t < KMAX; t += 32) {
    float m = kf[t] ? 1.f : 0.f;
    const float* g = gbox + ((size_t)b * KMAX + t) * 4;
    float* pp = props + ((size_t)b * KMAX + t) * 4;
    pp[0] = g[0] * m; pp[1] = g[1] * m; pp[2] = g[2] * m; pp[3] = g[3] * m;
    sco[(size_t)b * KMAX + t] = kf[t] ? gsc[(size_t)b * KMAX + t] : 0.f;
    kee[(size_t)b * KMAX + t] = m;
  }
}

// ---------------------------------------------------------------------------
extern "C" void kernel_launch(void* const* d_in, const int* in_sizes, int n_in,
                              void* d_out, int out_size, void* d_ws, size_t ws_size,
                              hipStream_t stream) {
  (void)in_sizes; (void)n_in; (void)out_size; (void)ws_size;
  const float* feat   = (const float*)d_in[1];
  const float* conv_w = (const float*)d_in[2];
  const float* conv_b = (const float*)d_in[3];
  const float* cls_w  = (const float*)d_in[4];
  const float* cls_b  = (const float*)d_in[5];
  const float* reg_w  = (const float*)d_in[6];
  const float* reg_b  = (const float*)d_in[7];

  char* ws = (char*)d_ws;
  size_t off = 0;
  auto alloc = [&](size_t bytes) -> void* {
    off = (off + 255) & ~(size_t)255;
    void* p = ws + off; off += bytes; return p;
  };
  unsigned short*     fn    = (unsigned short*)alloc((size_t)NPIX * CC * 2);
  unsigned short*     wt    = (unsigned short*)alloc((size_t)9 * COUT * CC * 2);
  unsigned short*     xo    = (unsigned short*)alloc((size_t)NPIX * COUT * 2);
  float*              cls_o = (float*)alloc((size_t)NPIX * 15 * 4);
  float*              reg_o = (float*)alloc((size_t)NPIX * 60 * 4);
  float*              boxes = (float*)alloc((size_t)BB * NANCH * 4 * 4);
  unsigned long long* keys  = (unsigned long long*)alloc((size_t)BB * NSORT * 8);
  float*              gbox  = (float*)alloc((size_t)BB * KMAX * 4 * 4);
  float*              gsc   = (float*)alloc((size_t)BB * KMAX * 4);
  unsigned int*       gval  = (unsigned int*)alloc((size_t)BB * KMAX * 4);
  unsigned int*       maskm = (unsigned int*)alloc((size_t)BB * KMAX * NWORD * 4);

  nhwc_kernel<<<(NPIX * CC) / 256, 256, 0, stream>>>(feat, fn);
  wt_kernel<<<(9 * COUT * CC) / 256, 256, 0, stream>>>(conv_w, wt);
  conv3x3_wmma<<<dim3(NPIX / 64, COUT / 128), 256, 0, stream>>>(fn, wt, conv_b, xo);
  heads_kernel<<<NPIX / 16, 256, 0, stream>>>(xo, cls_w, cls_b, reg_w, reg_b, cls_o, reg_o);
  decode_kernel<<<(BB * NSORT) / 256, 256, 0, stream>>>(cls_o, reg_o, boxes, keys);
  sort_kernel<<<BB, 1024, 0, stream>>>(keys);
  gather_kernel<<<BB, 256, 0, stream>>>(keys, boxes, gbox, gsc, gval);
  iou_kernel<<<dim3((KMAX * NWORD + 255) / 256, BB), 256, 0, stream>>>(gbox, maskm);
  nms_reduce_kernel<<<BB, 32, 0, stream>>>(maskm, gval, gbox, gsc, (float*)d_out);
}
